// LaneGNN_52664888983598
// MI455X (gfx1250) — matrix operand order repaired and checked by reading, not compiled, not run on hardware
//
#include <hip/hip_runtime.h>

typedef __attribute__((ext_vector_type(16))) _Float16 v16h;
typedef __attribute__((ext_vector_type(8)))  float    v8f;
typedef unsigned int u32x4 __attribute__((ext_vector_type(4)));

union ABPack { unsigned int u[8]; v16h h; };

#define NNODE 1024
#define NEDGE 8192
#define CROPS_PER_CHUNK 64
#define NCHUNKS 16

// ======================= generic WMMA GEMM =======================
// C[M x N] = A[M x K](f16,row-major, K = KT*32) * Wt[N x K](f16)^T + bias
// one wave per 16x16 tile; grid = (M/16, ceil(N/16)).
// A fragment per lane = two contiguous 16B runs; B fragment = two contiguous 16B runs.
template<int KT, int RELU, int OUTF16, int NGUARD>
__global__ __launch_bounds__(32) void k_gemm(
    const _Float16* __restrict__ A,
    const _Float16* __restrict__ Wt,
    const float* __restrict__ bias,
    void* __restrict__ out, int ldo, int N)
{
  constexpr int K = KT * 32;
  const int lane = threadIdx.x & 31;
  const int half = lane >> 4;
  const int rm   = lane & 15;
  const int mtile = blockIdx.x;
  const int ntile = blockIdx.y;
  const int m = mtile * 16 + rm;
  const int n = ntile * 16 + rm;
  bool nvalid = true;
  if constexpr (NGUARD) nvalid = (n < N);

  const _Float16* arow = A + (size_t)m * K;
  const _Float16* wrow = Wt + (size_t)(NGUARD ? (nvalid ? n : 0) : n) * K;

  v8f acc = {};
  for (int t = 0; t < KT; ++t) {
    const int k0 = t * 32;
    ABPack a, b;
    u32x4 alo = *(const u32x4*)(arow + k0 + half * 8);        // K: half*8 .. +7
    u32x4 ahi = *(const u32x4*)(arow + k0 + 16 + half * 8);   // K: 16+half*8 .. +7
    u32x4 w0  = *(const u32x4*)(wrow + k0 + half * 8);        // K: half*8 .. +7
    u32x4 w1  = *(const u32x4*)(wrow + k0 + 16 + half * 8);   // K: 16+half*8 .. +7
    if constexpr (NGUARD) {
      if (!nvalid) { w0 = (u32x4){0,0,0,0}; w1 = (u32x4){0,0,0,0}; }
    }
#pragma unroll
    for (int i = 0; i < 4; ++i) {
      a.u[i] = alo[i]; a.u[4 + i] = ahi[i];
      b.u[i] = w0[i];  b.u[4 + i] = w1[i];
    }
    acc = __builtin_amdgcn_wmma_f32_16x16x32_f16(false, a.h, false, b.h,
                                                 (short)0, acc, false, false);
  }

  const float bv = (NGUARD && !nvalid) ? 0.0f : bias[n];
#pragma unroll
  for (int r = 0; r < 8; ++r) {
    const int mm = mtile * 16 + r + half * 8;   // C/D: VGPR r -> M=r (+8 hi half)
    float v = acc[r] + bv;
    if constexpr (RELU) v = v > 0.0f ? v : 0.0f;
    if constexpr (NGUARD) { if (!nvalid) continue; }
    if constexpr (OUTF16) ((_Float16*)out)[(size_t)mm * ldo + n] = (_Float16)v;
    else                  ((float*)out)   [(size_t)mm * ldo + n] = v;
  }
}

// ======================= implicit-GEMM conv (stride 2, 3x3, SAME) =======================
// in : [crop][Hin*Win][Cin] f16 channels-last; Wt: [Cout][KPAD], k=(kh*3+kw)*Cin+cin
// out: [crop][Hout*Wout][Cout] f16, ReLU fused. grid=(crops*MTILES, COUT/16)
template<int HIN, int CINL2, int COUT, int KPAD, int MTILES>
__global__ __launch_bounds__(32) void k_conv(
    const _Float16* __restrict__ in,
    const _Float16* __restrict__ Wt,
    const float* __restrict__ bias,
    _Float16* __restrict__ out)
{
  constexpr int CIN  = 1 << CINL2;
  constexpr int HOUT = HIN >> 1;
  constexpr int MOUT = HOUT * HOUT;
  constexpr int KT   = KPAD / 32;
  constexpr int KTOT = 9 << CINL2;

  const int lane = threadIdx.x & 31;
  const int half = lane >> 4;
  const int rm   = lane & 15;
  const int crop  = blockIdx.x / MTILES;
  const int mtile = blockIdx.x % MTILES;
  const int m  = mtile * 16 + rm;
  const int oy = m / HOUT, ox = m % HOUT;
  const int n  = blockIdx.y * 16 + rm;

  const _Float16* inC  = in + (size_t)crop * HIN * HIN * CIN;
  const _Float16* wrow = Wt + (size_t)n * KPAD;

  v8f acc = {};
  for (int t = 0; t < KT; ++t) {
    const int k0 = t * 32;
    ABPack a, b;
    if constexpr (CINL2 >= 3) {
      // 8-element groups stay within one (kh,kw) patch -> one b128 load each
#pragma unroll
      for (int grp = 0; grp < 2; ++grp) {
        const int k   = k0 + grp * 16 + half * 8;
        const int p   = k >> CINL2;
        const int cin = k & (CIN - 1);
        const int kh = p / 3, kw = p % 3;
        const int iy = 2 * oy + kh, ix = 2 * ox + kw;   // SAME: pad_beg=0, pad_end=1
        u32x4 val = {0, 0, 0, 0};
        if (iy < HIN && ix < HIN)
          val = *(const u32x4*)(inC + (((size_t)(iy * HIN + ix)) << CINL2) + cin);
#pragma unroll
        for (int i = 0; i < 4; ++i) a.u[grp * 4 + i] = val[i];
      }
    } else {
      // conv0: Cin=4, gather dword pairs with K-pad guard
#pragma unroll
      for (int g = 0; g < 8; ++g) {
        const int kk = (g < 4) ? (half * 8 + 2 * g) : (16 + half * 8 + 2 * (g - 4));
        const int k  = k0 + kk;
        unsigned int val = 0u;
        if (k < KTOT) {
          const int p   = k >> CINL2;
          const int cin = k & (CIN - 1);
          const int kh = p / 3, kw = p % 3;
          const int iy = 2 * oy + kh, ix = 2 * ox + kw;
          if (iy < HIN && ix < HIN)
            val = *(const unsigned int*)(inC + (((size_t)(iy * HIN + ix)) << CINL2) + cin);
        }
        a.u[g] = val;
      }
    }
    u32x4 w0 = *(const u32x4*)(wrow + k0 + half * 8);
    u32x4 w1 = *(const u32x4*)(wrow + k0 + 16 + half * 8);
#pragma unroll
    for (int i = 0; i < 4; ++i) { b.u[i] = w0[i]; b.u[4 + i] = w1[i]; }
    acc = __builtin_amdgcn_wmma_f32_16x16x32_f16(false, a.h, false, b.h,
                                                 (short)0, acc, false, false);
  }

  const float bv = bias[n];
  _Float16* outc = out + (size_t)crop * MOUT * COUT;
#pragma unroll
  for (int r = 0; r < 8; ++r) {
    const int mm = mtile * 16 + r + half * 8;
    if constexpr (MOUT % 16 != 0) { if (mm >= MOUT) continue; }
    float v = acc[r] + bv;
    v = v > 0.0f ? v : 0.0f;
    outc[(size_t)mm * COUT + n] = (_Float16)v;
  }
}

// ======================= weight preparation =======================
__global__ void k_prep_lin(const float* __restrict__ W, _Float16* __restrict__ Wt,
                           int K, int N, int Kpad) {
  int idx = blockIdx.x * 256 + threadIdx.x;
  if (idx >= N * Kpad) return;
  int n = idx / Kpad, k = idx % Kpad;
  Wt[idx] = (k < K) ? (_Float16)W[(size_t)k * N + n] : (_Float16)0.0f;
}

__global__ void k_prep_conv(const float* __restrict__ W, _Float16* __restrict__ Wt,
                            int Cin, int Cout, int Kpad) {
  int idx = blockIdx.x * 256 + threadIdx.x;
  if (idx >= Cout * Kpad) return;
  int o = idx / Kpad, k = idx % Kpad;
  int K = 9 * Cin;
  _Float16 v = (_Float16)0.0f;
  if (k < K) {
    int p = k / Cin, cin = k % Cin;
    int kh = p / 3, kw = p % 3;
    v = (_Float16)W[(((size_t)o * Cin + cin) * 3 + kh) * 3 + kw];   // OIHW
  }
  Wt[idx] = v;
}

// map_W is [4608(=c*36+p)][128]; Wt[o][p*128+c] matches channels-last conv3 output
__global__ void k_prep_map(const float* __restrict__ W, _Float16* __restrict__ Wt) {
  int idx = blockIdx.x * 256 + threadIdx.x;
  if (idx >= 128 * 4608) return;
  int o = idx / 4608, k = idx % 4608;
  int p = k / 128, c = k % 128;
  Wt[idx] = (_Float16)W[((size_t)c * 36 + p) * 128 + o];
}

// ======================= input builders / gathers =======================
__global__ void k_pein(const int* __restrict__ nf, _Float16* __restrict__ out) {
  int idx = blockIdx.x * 256 + threadIdx.x;
  if (idx >= NNODE * 2) return;
  out[(idx >> 1) * 32 + (idx & 1)] = (_Float16)(float)nf[idx];
}

__global__ void k_eein(const float* __restrict__ ea4, _Float16* __restrict__ out) {
  int idx = blockIdx.x * 256 + threadIdx.x;
  if (idx >= NEDGE * 4) return;
  out[(idx >> 2) * 32 + (idx & 3)] = (_Float16)ea4[idx];
}

__global__ void k_copy16(const _Float16* __restrict__ s, _Float16* __restrict__ d, int n) {
  int idx = blockIdx.x * 256 + threadIdx.x;
  if (idx < n) d[idx] = s[idx];
}

__global__ void k_crops(const int* __restrict__ nf, const int* __restrict__ batch,
                        const float* __restrict__ rgb, const float* __restrict__ regr,
                        _Float16* __restrict__ out, int crop0) {
  int idx = blockIdx.x * 256 + threadIdx.x;            // local_crop*9216 + px
  if (idx >= CROPS_PER_CHUNK * 9216) return;
  int cl = idx / 9216, px = idx % 9216;
  int g = crop0 + cl;
  int yy = px / 96, xx = px % 96;
  int b = batch[g];
  int top = nf[g * 2 + 1] + 80, left = nf[g * 2 + 0] + 80;
  int iy = top + yy, ix = left + xx;
  _Float16* o = out + (size_t)idx * 4;
#pragma unroll
  for (int c = 0; c < 3; ++c)
    o[c] = (_Float16)rgb[(((size_t)b * 3 + c) * 512 + iy) * 512 + ix];
  o[3] = (_Float16)regr[(((size_t)b) * 512 + iy) * 512 + ix];
}

__global__ void k_fnin(const _Float16* __restrict__ x, const _Float16* __restrict__ nma,
                       _Float16* __restrict__ out) {
  int idx = blockIdx.x * 256 + threadIdx.x;
  if (idx >= NNODE * 192) return;
  int i = idx / 192, j = idx % 192;
  out[idx] = (j < 64) ? x[i * 64 + j] : nma[i * 128 + (j - 64)];
}

__global__ void k_concat_eu(const _Float16* __restrict__ x, const _Float16* __restrict__ ea,
                            const int* __restrict__ ei, _Float16* __restrict__ out) {
  int idx = blockIdx.x * 256 + threadIdx.x;
  if (idx >= NEDGE * 192) return;
  int e = idx / 192, j = idx % 192;
  _Float16 v;
  if (j < 64)       v = x[ei[e] * 64 + j];
  else if (j < 128) v = x[ei[NEDGE + e] * 64 + (j - 64)];
  else              v = ea[e * 64 + (j - 128)];
  out[idx] = v;
}

__global__ void k_concat_msg(const _Float16* __restrict__ x, const _Float16* __restrict__ ea,
                             const int* __restrict__ ei, _Float16* __restrict__ out) {
  int idx = blockIdx.x * 256 + threadIdx.x;
  if (idx >= NEDGE * 128) return;
  int e = idx / 128, j = idx % 128;
  out[idx] = (j < 64) ? x[ei[e] * 64 + j] : ea[e * 64 + (j - 64)];
}

__global__ void k_concat_nu(const _Float16* __restrict__ x, const float* __restrict__ agg,
                            const _Float16* __restrict__ x0, _Float16* __restrict__ out) {
  int idx = blockIdx.x * 256 + threadIdx.x;
  if (idx >= NNODE * 192) return;
  int i = idx / 192, j = idx % 192;
  _Float16 v;
  if (j < 64)       v = x[i * 64 + j];
  else if (j < 128) v = (_Float16)agg[i * 64 + (j - 64)];
  else              v = x0[i * 64 + (j - 128)];
  out[idx] = v;
}

// deterministic segment-sum: one block per node, dst[] cached in LDS, lane c reduces channel c
__global__ __launch_bounds__(64) void k_segsum(const float* __restrict__ m,
                                               const int* __restrict__ ei,
                                               float* __restrict__ agg) {
  __shared__ int sdst[NEDGE];
  int node = blockIdx.x;
  for (int e = threadIdx.x; e < NEDGE; e += 64) sdst[e] = ei[NEDGE + e];
  __syncthreads();
  int c = threadIdx.x;
  float s = 0.0f;
  for (int e = 0; e < NEDGE; ++e)
    if (sdst[e] == node) s += m[(size_t)e * 64 + c];
  agg[node * 64 + c] = s;
}

// ======================= host orchestration =======================
template<int KT, int RELU, int OUTF16, int NGUARD>
static inline void gemmT(hipStream_t s, const _Float16* A, const _Float16* Wt,
                         const float* bias, void* out, int ldo, int M, int N) {
  k_gemm<KT, RELU, OUTF16, NGUARD><<<dim3(M / 16, (N + 15) / 16), dim3(32), 0, s>>>(
      A, Wt, bias, out, ldo, N);
}

extern "C" void kernel_launch(void* const* d_in, const int* in_sizes, int n_in,
                              void* d_out, int out_size, void* d_ws, size_t ws_size,
                              hipStream_t stream) {
  (void)in_sizes; (void)n_in; (void)out_size; (void)ws_size;
  const int*   nf    = (const int*)d_in[0];
  const float* ea4   = (const float*)d_in[1];
  const int*   ei    = (const int*)d_in[2];
  const int*   batch = (const int*)d_in[3];
  const float* rgb   = (const float*)d_in[4];
  const float* regr  = (const float*)d_in[5];
  const float* const* P = (const float* const*)(d_in + 6);   // params, sorted-key order
  enum {C0_B=0,C0_K,C1_B,C1_K,C2_B,C2_K,C3_B,C3_K,
        EC_W1,EC_W2,EC_W3,EC_B1,EC_B2,EC_B3,
        EE_W1,EE_W2,EE_B1,EE_B2,
        EPC_W1,EPC_W2,EPC_W3,EPC_B1,EPC_B2,EPC_B3,
        EU_W1,EU_W2,EU_B1,EU_B2,
        FN_W1,FN_W2,FN_B1,FN_B2,
        MAP_W,MAP_B,
        MSG_W1,MSG_W2,MSG_B1,MSG_B2,
        NC_W1,NC_W2,NC_W3,NC_B1,NC_B2,NC_B3,
        NU_W1,NU_W2,NU_B1,NU_B2,
        PE_W1,PE_W2,PE_B1,PE_B2};

  size_t cur = 0;
  auto alloc = [&](size_t bytes) -> char* {
    char* p = (char*)d_ws + cur;
    cur += (bytes + 255) & ~(size_t)255;
    return p;
  };
  auto aH = [&](size_t elems) -> _Float16* { return (_Float16*)alloc(elems * 2); };
  auto aF = [&](size_t elems) -> float*    { return (float*)alloc(elems * 4); };

  // transposed/padded f16 weights [N][Kpad]
  _Float16 *peW1t = aH(32*32),   *peW2t = aH(64*32);
  _Float16 *eeW1t = aH(32*32),   *eeW2t = aH(64*32);
  _Float16 *fnW1t = aH(128*192), *fnW2t = aH(64*128);
  _Float16 *euW1t = aH(64*192),  *euW2t = aH(64*64);
  _Float16 *msW1t = aH(64*128),  *msW2t = aH(64*64);
  _Float16 *nuW1t = aH(64*192),  *nuW2t = aH(64*64);
  _Float16 *ecW1t = aH(32*64),   *ecW2t = aH(16*32), *ecW3t = aH(1*32);
  _Float16 *ncW1t = aH(32*64),   *ncW2t = aH(16*32), *ncW3t = aH(1*32);
  _Float16 *epW1t = aH(32*64),   *epW2t = aH(16*32), *epW3t = aH(1*32);
  _Float16 *mapWt = aH(128*4608);
  _Float16 *c0Wt = aH(32*64), *c1Wt = aH(64*288), *c2Wt = aH(128*576), *c3Wt = aH(128*1152);

  // activations
  _Float16 *pe_in = aH(NNODE*32),  *h_pe = aH(NNODE*32);
  _Float16 *ee_in = aH(NEDGE*32),  *h_ee = aH(NEDGE*32);
  _Float16 *x  = aH(NNODE*64), *x0 = aH(NNODE*64);
  _Float16 *nma = aH(NNODE*128);
  _Float16 *ea  = aH(NEDGE*64);
  _Float16 *fn_in = aH(NNODE*192), *fn_h = aH(NNODE*128);
  _Float16 *eu_in = aH(NEDGE*192), *eu_h = aH(NEDGE*64);
  _Float16 *ms_in = aH(NEDGE*128), *ms_h = aH(NEDGE*64);
  float    *m_f32 = aF(NEDGE*64),  *agg = aF(NNODE*64);
  _Float16 *nu_in = aH(NNODE*192), *nu_h = aH(NNODE*64);
  _Float16 *hc1 = aH(NEDGE*32),    *hc2 = aH(NEDGE*32);
  // conv chunk buffers (channels-last f16)
  _Float16 *cact = aH((size_t)CROPS_PER_CHUNK*9216*4);
  _Float16 *c0o  = aH((size_t)CROPS_PER_CHUNK*2304*32);
  _Float16 *c1o  = aH((size_t)CROPS_PER_CHUNK*576*64);
  _Float16 *c2o  = aH((size_t)CROPS_PER_CHUNK*144*128);
  _Float16 *c3o  = aH((size_t)CROPS_PER_CHUNK*36*128);

  auto prep = [&](const float* W, _Float16* Wt, int K, int N, int Kpad) {
    int tot = N * Kpad;
    k_prep_lin<<<dim3((tot + 255) / 256), dim3(256), 0, stream>>>(W, Wt, K, N, Kpad);
  };

  // ---- weight prep ----
  prep(P[PE_W1], peW1t, 2, 32, 32);    prep(P[PE_W2], peW2t, 32, 64, 32);
  prep(P[EE_W1], eeW1t, 4, 32, 32);    prep(P[EE_W2], eeW2t, 32, 64, 32);
  prep(P[FN_W1], fnW1t, 192, 128, 192);prep(P[FN_W2], fnW2t, 128, 64, 128);
  prep(P[EU_W1], euW1t, 192, 64, 192); prep(P[EU_W2], euW2t, 64, 64, 64);
  prep(P[MSG_W1], msW1t, 128, 64, 128);prep(P[MSG_W2], msW2t, 64, 64, 64);
  prep(P[NU_W1], nuW1t, 192, 64, 192); prep(P[NU_W2], nuW2t, 64, 64, 64);
  prep(P[EC_W1], ecW1t, 64, 32, 64);   prep(P[EC_W2], ecW2t, 32, 16, 32);  prep(P[EC_W3], ecW3t, 16, 1, 32);
  prep(P[NC_W1], ncW1t, 64, 32, 64);   prep(P[NC_W2], ncW2t, 32, 16, 32);  prep(P[NC_W3], ncW3t, 16, 1, 32);
  prep(P[EPC_W1], epW1t, 64, 32, 64);  prep(P[EPC_W2], epW2t, 32, 16, 32); prep(P[EPC_W3], epW3t, 16, 1, 32);
  k_prep_map<<<dim3((128*4608 + 255)/256), dim3(256), 0, stream>>>(P[MAP_W], mapWt);
  k_prep_conv<<<dim3((32*64 + 255)/256),    dim3(256), 0, stream>>>(P[C0_K], c0Wt, 4, 32, 64);
  k_prep_conv<<<dim3((64*288 + 255)/256),   dim3(256), 0, stream>>>(P[C1_K], c1Wt, 32, 64, 288);
  k_prep_conv<<<dim3((128*576 + 255)/256),  dim3(256), 0, stream>>>(P[C2_K], c2Wt, 64, 128, 576);
  k_prep_conv<<<dim3((128*1152 + 255)/256), dim3(256), 0, stream>>>(P[C3_K], c3Wt, 128, 128, 1152);

  // ---- zero-padded input regions ----
  hipMemsetAsync(pe_in, 0, (size_t)NNODE * 32 * 2, stream);
  hipMemsetAsync(ee_in, 0, (size_t)NEDGE * 32 * 2, stream);
  hipMemsetAsync(hc2,   0, (size_t)NEDGE * 32 * 2, stream);

  // ---- node / edge encoders ----
  k_pein<<<dim3((NNODE*2 + 255)/256), dim3(256), 0, stream>>>(nf, pe_in);
  gemmT<1,1,1,0>(stream, pe_in, peW1t, P[PE_B1], h_pe, 32, NNODE, 32);
  gemmT<1,0,1,0>(stream, h_pe, peW2t, P[PE_B2], x, 64, NNODE, 64);
  k_copy16<<<dim3((NNODE*64 + 255)/256), dim3(256), 0, stream>>>(x, x0, NNODE*64);

  k_eein<<<dim3((NEDGE*4 + 255)/256), dim3(256), 0, stream>>>(ea4, ee_in);
  gemmT<1,1,1,0>(stream, ee_in, eeW1t, P[EE_B1], h_ee, 32, NEDGE, 32);
  gemmT<1,0,1,0>(stream, h_ee, eeW2t, P[EE_B2], ea, 64, NEDGE, 64);

  // ---- crop extraction + conv stack + map projection, chunked ----
  for (int ch = 0; ch < NCHUNKS; ++ch) {
    int crop0 = ch * CROPS_PER_CHUNK;
    k_crops<<<dim3((CROPS_PER_CHUNK*9216 + 255)/256), dim3(256), 0, stream>>>(
        nf, batch, rgb, regr, cact, crop0);
    k_conv<96, 2, 32, 64, 144><<<dim3(CROPS_PER_CHUNK*144, 2), dim3(32), 0, stream>>>(cact, c0Wt, P[C0_B], c0o);
    k_conv<48, 5, 64, 288, 36><<<dim3(CROPS_PER_CHUNK*36, 4),  dim3(32), 0, stream>>>(c0o,  c1Wt, P[C1_B], c1o);
    k_conv<24, 6, 128, 576, 9><<<dim3(CROPS_PER_CHUNK*9, 8),   dim3(32), 0, stream>>>(c1o,  c2Wt, P[C2_B], c2o);
    k_conv<12, 7, 128, 1152, 3><<<dim3(CROPS_PER_CHUNK*3, 8),  dim3(32), 0, stream>>>(c2o,  c3Wt, P[C3_B], c3o);
    // flatten (already [crop][4608]) -> map_W (Wt permuted to channels-last K-order)
    gemmT<144,0,1,0>(stream, c3o, mapWt, P[MAP_B],
                     (void*)(nma + (size_t)crop0 * 128), 128, CROPS_PER_CHUNK, 128);
  }

  // ---- fuse map features into node state ----
  k_fnin<<<dim3((NNODE*192 + 255)/256), dim3(256), 0, stream>>>(x, nma, fn_in);
  gemmT<6,1,1,0>(stream, fn_in, fnW1t, P[FN_B1], fn_h, 128, NNODE, 128);
  gemmT<4,0,1,0>(stream, fn_h, fnW2t, P[FN_B2], x, 64, NNODE, 64);

  // ---- 8 rounds of message passing ----
  for (int d = 0; d < 8; ++d) {
    k_concat_eu<<<dim3((NEDGE*192 + 255)/256), dim3(256), 0, stream>>>(x, ea, ei, eu_in);
    gemmT<6,1,1,0>(stream, eu_in, euW1t, P[EU_B1], eu_h, 64, NEDGE, 64);
    gemmT<2,0,1,0>(stream, eu_h, euW2t, P[EU_B2], ea, 64, NEDGE, 64);

    k_concat_msg<<<dim3((NEDGE*128 + 255)/256), dim3(256), 0, stream>>>(x, ea, ei, ms_in);
    gemmT<4,1,1,0>(stream, ms_in, msW1t, P[MSG_B1], ms_h, 64, NEDGE, 64);
    gemmT<2,0,0,0>(stream, ms_h, msW2t, P[MSG_B2], m_f32, 64, NEDGE, 64);

    k_segsum<<<dim3(NNODE), dim3(64), 0, stream>>>(m_f32, ei, agg);

    k_concat_nu<<<dim3((NNODE*192 + 255)/256), dim3(256), 0, stream>>>(x, agg, x0, nu_in);
    gemmT<6,1,1,0>(stream, nu_in, nuW1t, P[NU_B1], nu_h, 64, NNODE, 64);
    gemmT<2,0,1,0>(stream, nu_h, nuW2t, P[NU_B2], x, 64, NNODE, 64);
  }

  // ---- heads: ec(edge), nc(node), epc(node) -> f32 outputs, concatenated ----
  float* out_ec  = (float*)d_out;
  float* out_nc  = out_ec + NEDGE;
  float* out_epc = out_nc + NNODE;

  gemmT<2,1,1,0>(stream, ea, ecW1t, P[EC_B1], hc1, 32, NEDGE, 32);
  gemmT<1,1,1,0>(stream, hc1, ecW2t, P[EC_B2], hc2, 32, NEDGE, 16);
  gemmT<1,0,0,1>(stream, hc2, ecW3t, P[EC_B3], out_ec, 1, NEDGE, 1);

  gemmT<2,1,1,0>(stream, x, ncW1t, P[NC_B1], hc1, 32, NNODE, 32);
  gemmT<1,1,1,0>(stream, hc1, ncW2t, P[NC_B2], hc2, 32, NNODE, 16);
  gemmT<1,0,0,1>(stream, hc2, ncW3t, P[NC_B3], out_nc, 1, NNODE, 1);

  gemmT<2,1,1,0>(stream, x, epW1t, P[EPC_B1], hc1, 32, NNODE, 32);
  gemmT<1,1,1,0>(stream, hc1, epW2t, P[EPC_B2], hc2, 32, NNODE, 16);
  gemmT<1,0,0,1>(stream, hc2, epW3t, P[EPC_B3], out_epc, 1, NNODE, 1);
}
